// ConvLSTMBlock_14061722927501
// MI455X (gfx1250) — matrix operand order repaired and checked by reading, not compiled, not run on hardware
//
#include <hip/hip_runtime.h>

typedef __attribute__((ext_vector_type(16))) __bf16 v16bf;
typedef __attribute__((ext_vector_type(8)))  float  v8f;

#define NV   7
#define NB   8
#define NS   10
#define NF   32
#define NCH  33          // 1 (x) + 32 (h)
#define CPAD 64          // padded channels: 2 x K=32 WMMA chunks
#define OC   128         // 4*NF gate channels
#define H    48
#define W    48
#define HW   (H*W)
#define NN   (NV*NB)     // 56 folded samples
#define OUT_ELEMS  ((size_t)NB*NS*NV*NF*HW)            // 41,287,680 f32 (hidden states)
// workspace byte offsets
#define WS_PACKW   0                                   // 144 frags * 512 u16 = 147,456 B
#define WS_XCOMB   (147456)                            // NN*HW*CPAD u16     = 16,515,072 B
#define WS_CSTATE  (147456 + 16515072)                 // NN*NF*HW f32       = 16,515,072 B

__device__ __forceinline__ unsigned short f32_to_bf16(float f) {
    union { float f; unsigned u; } v; v.f = f;
    unsigned r = v.u + 0x7FFFu + ((v.u >> 16) & 1u);   // round-to-nearest-even
    return (unsigned short)(r >> 16);
}

// ---------------------------------------------------------------------------
// Pack Wk (128,33,3,3) f32 -> bf16 A-fragments in the exact wave32 16-bit
// A-matrix (16x32) VGPR layout, per (oc-tile mt, tap, K-chunk q).
// Lane l, element j -> M = l&15 ; K = (j>>3)*16 + (l>>4)*8 + (j&7).
// ---------------------------------------------------------------------------
__global__ void pack_weights(const float* __restrict__ Wk, unsigned short* __restrict__ pw) {
    int idx = blockIdx.x * 256 + threadIdx.x;
    if (idx >= 144 * 512) return;
    int frag = idx >> 9;                 // 0..143 = ((mt*9 + tap)*2 + q)
    int l    = (idx >> 4) & 31;
    int j    = idx & 15;
    int q    = frag & 1;
    int tap  = (frag >> 1) % 9;
    int mt   = frag / 18;
    int K    = ((j >> 3) << 4) + ((l >> 4) << 3) + (j & 7);
    int ci   = q * 32 + K;               // combined-input channel (0 = x, 1..32 = h)
    int oc   = mt * 16 + (l & 15);
    float w  = (ci < NCH) ? Wk[(oc * NCH + ci) * 9 + tap] : 0.0f;
    pw[idx]  = f32_to_bf16(w);
}

// ---------------------------------------------------------------------------
// Init combined input tensor xc[n][y][x][c64] (bf16): zeros, ch0 = x at t=0.
// ---------------------------------------------------------------------------
__global__ void setup_x(const float* __restrict__ in, unsigned short* __restrict__ xc) {
    int p = blockIdx.x * 256 + threadIdx.x;          // (n, hw)
    if (p >= NN * HW) return;
    int n = p / HW, hw = p % HW;
    int v = n >> 3, b = n & 7;
    unsigned short* dst = xc + (size_t)p * CPAD;
    uint4 z = {0u, 0u, 0u, 0u};
#pragma unroll
    for (int c = 0; c < 8; ++c) ((uint4*)dst)[c] = z;
    dst[0] = f32_to_bf16(in[((b * NS + 0) * NV + v) * HW + hw]);
}

__global__ void zero_c(float* __restrict__ c) {
    int i = blockIdx.x * 256 + threadIdx.x;
    if (i < NN * NF * HW) c[i] = 0.0f;
}

// ---------------------------------------------------------------------------
// gates = conv3x3([x;h;pad], Wk) + bk  as implicit GEMM on bf16 WMMA.
// Block: 128 oc x 16 x-positions of one (n, y) row. 8 waves, 1 oc-16-tile each.
// LDS halo tile: 3 rows x 18 cols x 64 ch, position stride padded to 72 u16
// (conflict-free ds_load_b128 B-fragment reads).
// ---------------------------------------------------------------------------
__global__ __launch_bounds__(256)
void gates_gemm(const unsigned short* __restrict__ xc,
                const unsigned short* __restrict__ pw,
                const float* __restrict__ bk,
                float* __restrict__ gates, int t) {
    __shared__ unsigned short tile[3 * 18 * 72];     // 7776 B

    const int x0  = blockIdx.x * 16;
    const int y   = blockIdx.y;
    const int n   = blockIdx.z;
    const int tid = threadIdx.x;

    // cooperative halo stage: 54 positions * 128 B, zero-filled at borders
    for (int e = tid; e < 54 * 8; e += 256) {
        int pi = e >> 3, cj = e & 7;
        int row = pi / 18, cc = pi % 18;
        int yy = y + row - 1, xx = x0 + cc - 1;
        uint4 val = {0u, 0u, 0u, 0u};
        if (yy >= 0 && yy < H && xx >= 0 && xx < W)
            val = *(const uint4*)(xc + ((size_t)(n * HW + yy * W + xx)) * CPAD + cj * 8);
        *(uint4*)(&tile[row * (18 * 72) + cc * 72 + cj * 8]) = val;
    }
    __syncthreads();

    const int wave = tid >> 5, lane = tid & 31;
    const int col  = lane & 15, hi = lane >> 4;

    v8f acc = {};
    const unsigned short* wbase = pw + (size_t)wave * 18 * 512 + lane * 16;

#pragma unroll
    for (int tap = 0; tap < 9; ++tap) {
        const int ky = tap / 3, kx = tap % 3;
#pragma unroll
        for (int q = 0; q < 2; ++q) {
            union { v16bf v; uint4 u[2]; } Af, Bf;
            const uint4* ap = (const uint4*)(wbase + (tap * 2 + q) * 512);
            Af.u[0] = ap[0];
            Af.u[1] = ap[1];
            const unsigned short* bp = &tile[ky * (18 * 72) + (col + kx) * 72 + q * 32 + hi * 8];
            Bf.u[0] = *(const uint4*)bp;
            Bf.u[1] = *(const uint4*)(bp + 16);
            acc = __builtin_amdgcn_wmma_f32_16x16x32_bf16(
                false, Af.v, false, Bf.v, (short)0, acc, false, false);
        }
    }

    // bias + store raw gates into gates_all[(v, t, b, oc, y, x)]
    const int vv = n >> 3, b = n & 7;
    const long gbase = ((((long)vv * NS + t) * NB + b) * OC) * HW + (long)y * W + x0 + col;
    const int ocb = wave * 16 + hi * 8;
#pragma unroll
    for (int r = 0; r < 8; ++r) {
        int oc = ocb + r;
        gates[gbase + (long)oc * HW] = acc[r] + bk[oc];
    }
}

// ---------------------------------------------------------------------------
// LSTM pointwise: cy = sig(f)*c + sig(i)*cg ; hy = sig(o)*tanh(cy)
// Writes hy (f32) to d_out, cy to ws, hy/x_{t+1} (bf16) into combined tensor.
// ---------------------------------------------------------------------------
__global__ void lstm_pw(const float* __restrict__ gates, float* __restrict__ cst,
                        const float* __restrict__ in, unsigned short* __restrict__ xc,
                        float* __restrict__ out, int t) {
    int e = blockIdx.x * 256 + threadIdx.x;
    if (e >= NN * NF * HW) return;
    int n  = e / (NF * HW);
    int rf = e % (NF * HW);
    int f  = rf / HW, hw = rf % HW;
    int v  = n >> 3, b = n & 7;

    long gb = ((((long)v * NS + t) * NB + b) * OC) * HW + hw;
    float ig = gates[gb + (long)(0 * NF + f) * HW];
    float fg = gates[gb + (long)(1 * NF + f) * HW];
    float cg = gates[gb + (long)(2 * NF + f) * HW];
    float og = gates[gb + (long)(3 * NF + f) * HW];
    ig = 1.0f / (1.0f + __expf(-ig));
    fg = 1.0f / (1.0f + __expf(-fg));
    og = 1.0f / (1.0f + __expf(-og));

    float cy = fg * cst[e] + ig * cg;       // NOTE: no tanh on cellgate (faithful)
    float hy = og * tanhf(cy);
    cst[e] = cy;

    out[(((long)b * NS + t) * (NV * NF) + v * NF + f) * HW + hw] = hy;
    xc[((size_t)n * HW + hw) * CPAD + 1 + f] = f32_to_bf16(hy);
    if (f == 0 && t + 1 < NS) {
        float xn = in[((b * NS + (t + 1)) * NV + v) * HW + hw];
        xc[((size_t)n * HW + hw) * CPAD + 0] = f32_to_bf16(xn);
    }
}

// ---------------------------------------------------------------------------
extern "C" void kernel_launch(void* const* d_in, const int* in_sizes, int n_in,
                              void* d_out, int out_size, void* d_ws, size_t ws_size,
                              hipStream_t stream) {
    const float* inputs = (const float*)d_in[0];
    const float* Wk     = (const float*)d_in[1];
    const float* bk     = (const float*)d_in[2];
    float* out   = (float*)d_out;
    float* gates = out + OUT_ELEMS;                      // gates_all region of d_out
    unsigned short* pw  = (unsigned short*)((char*)d_ws + WS_PACKW);
    unsigned short* xcm = (unsigned short*)((char*)d_ws + WS_XCOMB);
    float* cst          = (float*)((char*)d_ws + WS_CSTATE);

    pack_weights<<<(144 * 512 + 255) / 256, 256, 0, stream>>>(Wk, pw);
    setup_x<<<(NN * HW + 255) / 256, 256, 0, stream>>>(inputs, xcm);
    zero_c<<<(NN * NF * HW + 255) / 256, 256, 0, stream>>>(cst);

    for (int t = 0; t < NS; ++t) {
        gates_gemm<<<dim3(3, H, NN), 256, 0, stream>>>(xcm, pw, bk, gates, t);
        lstm_pw<<<(NN * NF * HW + 255) / 256, 256, 0, stream>>>(gates, cst, inputs, xcm, out, t);
    }
}
